// LinearBlock_17927193493941
// MI455X (gfx1250) — 8-bit matrix operand order repaired and checked by reading, compiled, not run on hardware
//
#include <hip/hip_runtime.h>


#define GM 8192
#define GK 4096
#define GN 4096

typedef __attribute__((ext_vector_type(8))) int       v8i;
typedef __attribute__((ext_vector_type(4))) int       v4i;
typedef __attribute__((ext_vector_type(4))) unsigned  u32x4;
typedef __attribute__((ext_vector_type(8))) int       i32x8;
typedef __attribute__((ext_vector_type(4))) int       i32x4;

union V8 { v8i v; v4i q[2]; long l[4]; int i[8]; };

__global__ void init_scales_kernel(unsigned* scales) {
  if (threadIdx.x == 0) { scales[0] = 0u; scales[1] = 0u; }
}

__global__ void absmax_kernel(const float* __restrict__ v, size_t n4,
                              unsigned* __restrict__ out) {
  size_t i = (size_t)blockIdx.x * blockDim.x + threadIdx.x;
  const size_t stride = (size_t)gridDim.x * blockDim.x;
  const float4* v4 = (const float4*)v;
  float m = 0.0f;
  for (; i < n4; i += stride) {
    float4 a = v4[i];
    m = fmaxf(m, fmaxf(fmaxf(fabsf(a.x), fabsf(a.y)),
                       fmaxf(fabsf(a.z), fabsf(a.w))));
  }
  #pragma unroll
  for (int off = 16; off > 0; off >>= 1)
    m = fmaxf(m, __shfl_xor(m, off, 32));
  if ((threadIdx.x & 31) == 0)
    atomicMax(out, __float_as_uint(m));
}

__global__ void quant_x_kernel(const float* __restrict__ v,
                               unsigned* __restrict__ q,
                               const unsigned* __restrict__ scale_bits,
                               size_t n4) {
  const float alpha = __uint_as_float(*scale_bits);
  const float s = 127.0f / alpha;
  size_t i = (size_t)blockIdx.x * blockDim.x + threadIdx.x;
  if (i >= n4) return;
  float4 a = ((const float4*)v)[i];
  int q0 = (int)rintf(fminf(fmaxf(a.x * s, -127.0f), 127.0f));
  int q1 = (int)rintf(fminf(fmaxf(a.y * s, -127.0f), 127.0f));
  int q2 = (int)rintf(fminf(fmaxf(a.z * s, -127.0f), 127.0f));
  int q3 = (int)rintf(fminf(fmaxf(a.w * s, -127.0f), 127.0f));
  q[i] = (unsigned)(q0 & 255) | ((unsigned)(q1 & 255) << 8) |
         ((unsigned)(q2 & 255) << 16) | ((unsigned)(q3 & 255) << 24);
}

__global__ void quant_wT_kernel(const float* __restrict__ W,
                                unsigned char* __restrict__ QwT,
                                const unsigned* __restrict__ scale_bits) {
  __shared__ char tile[64 * 68];
  const float alpha = __uint_as_float(*scale_bits);
  const float s = 127.0f / alpha;
  const int k0 = blockIdx.x * 64;
  const int n0 = blockIdx.y * 64;
  const int t  = threadIdx.x;

  const int kl = t & 63;
  const int nb = t >> 6;
  #pragma unroll
  for (int i = 0; i < 16; ++i) {
    const int nl = nb + i * 4;
    float a = W[(size_t)(n0 + nl) * GK + k0 + kl];
    int qv = (int)rintf(fminf(fmaxf(a * s, -127.0f), 127.0f));
    tile[kl * 68 + nl] = (char)qv;
  }
  __syncthreads();
  const int nl = t & 63;
  const int kb = t >> 6;
  #pragma unroll
  for (int i = 0; i < 16; ++i) {
    const int kl2 = kb + i * 4;
    QwT[(size_t)(k0 + kl2) * GN + n0 + nl] = (unsigned char)tile[kl2 * 68 + nl];
  }
}

__device__ __forceinline__ void tdm_load_2d(unsigned long long gaddr,
                                            unsigned lds_byte_addr,
                                            i32x8 g1) {
  u32x4 g0;
  g0.x = 1u;
  g0.y = lds_byte_addr;
  g0.z = (unsigned)(gaddr & 0xffffffffull);
  g0.w = (unsigned)(gaddr >> 32) | (2u << 30);
  const i32x4 z4 = {0, 0, 0, 0};
  const i32x8 z8 = {0, 0, 0, 0, 0, 0, 0, 0};
  __builtin_amdgcn_tensor_load_to_lds(g0, g1, z4, z4, z8, 0);
}

__device__ __forceinline__ i32x8 tdm_group1_A() {
  i32x8 g;
  g[0] = (1 << 20) | (3 << 22) | (1 << 25);
  g[1] = (int)(4096u << 16);
  g[2] = (int)(8192u << 16);
  g[3] = (64 << 16);
  g[4] = 128;
  g[5] = 4096;
  g[6] = 0;
  g[7] = 0;
  return g;
}

__device__ __forceinline__ i32x8 tdm_group1_B() {
  i32x8 g;
  g[0] = (1 << 20) | (4 << 22) | (3 << 25);
  g[1] = (int)(4096u << 16);
  g[2] = (int)(4096u << 16);
  g[3] = (128 << 16);
  g[4] = 64;
  g[5] = 4096;
  g[6] = 0;
  g[7] = 0;
  return g;
}

#define A_BUF_BYTES (128 * 72)
#define B_BUF_BYTES (64 * 144)
#define B_LDS_BASE  (2 * A_BUF_BYTES)
#define SMEM_BYTES  (2 * A_BUF_BYTES + 2 * B_BUF_BYTES)

__global__ __launch_bounds__(256)
void gemm_i8_wmma_kernel(const unsigned char* __restrict__ Qx,
                         const unsigned char* __restrict__ QwT,
                         float* __restrict__ out,
                         const unsigned* __restrict__ scales) {
  extern __shared__ __align__(16) unsigned char smem[];

  const int t    = threadIdx.x;
  const int lane = t & 31;
  const int wave = t >> 5;
  const int wm   = wave & 1;
  const int wn   = wave >> 1;
  const int bm   = blockIdx.x * 128;
  const int bn   = blockIdx.y * 128;

  const unsigned long long gA = (unsigned long long)(size_t)Qx + (size_t)bm * GK;
  const unsigned long long gB = (unsigned long long)(size_t)QwT + (size_t)bn;

  const v8i vzero = {0, 0, 0, 0, 0, 0, 0, 0};
  v8i acc[4][2];
  #pragma unroll
  for (int i = 0; i < 4; ++i)
    #pragma unroll
    for (int j = 0; j < 2; ++j) acc[i][j] = vzero;

  const i32x8 g1a = tdm_group1_A();
  const i32x8 g1b = tdm_group1_B();

  if (wave == 0) {
    tdm_load_2d(gA, 0u, g1a);
    __builtin_amdgcn_s_wait_tensorcnt(0);
  } else if (wave == 1) {
    tdm_load_2d(gB, B_LDS_BASE, g1b);
    __builtin_amdgcn_s_wait_tensorcnt(0);
  }
  __syncthreads();

  const int NKT = GK / 64;
  #pragma unroll 1
  for (int kt = 0; kt < NKT; ++kt) {
    const int cur = kt & 1;
    const int nxt = cur ^ 1;
    const bool more = (kt + 1 < NKT);
    if (more && wave == 0)
      tdm_load_2d(gA + (size_t)(kt + 1) * 64, (unsigned)(nxt * A_BUF_BYTES), g1a);
    if (more && wave == 1)
      tdm_load_2d(gB + (size_t)(kt + 1) * 64 * GN,
                  (unsigned)(B_LDS_BASE + nxt * B_BUF_BYTES), g1b);

    const unsigned char* Acur = smem + cur * A_BUF_BYTES;
    const unsigned char* Bcur = smem + B_LDS_BASE + cur * B_BUF_BYTES;

    V8 bf[2];
    #pragma unroll
    for (int ns = 0; ns < 2; ++ns) {
      const int nb = wn * 32 + ns * 16;
      #pragma unroll
      for (int w = 0; w < 8; ++w) {
        int v = 0;
        #pragma unroll
        for (int b = 0; b < 4; ++b) {
          const int s = w * 4 + b;
          const int kk = 16 * (s >> 3) + (lane >> 4) * 8 + (s & 7);
          v |= (int)Bcur[kk * 144 + nb + (lane & 15)] << (8 * b);
        }
        bf[ns].i[w] = v;
      }
    }

    const int o = (lane >> 4) * 8;
    #pragma unroll
    for (int ms = 0; ms < 4; ++ms) {
      const int M = wm * 64 + ms * 16 + (lane & 15);
      const unsigned char* arow = Acur + M * 72 + o;
      V8 af;
      af.l[0] = *(const long*)(arow);
      af.l[1] = *(const long*)(arow + 16);
      af.l[2] = *(const long*)(arow + 32);
      af.l[3] = *(const long*)(arow + 48);
      #pragma unroll
      for (int ns = 0; ns < 2; ++ns)
        acc[ms][ns] = __builtin_amdgcn_wmma_i32_16x16x64_iu8(
            true, af.v, true, bf[ns].v, acc[ms][ns], false, false);
    }

    if (more && wave < 2)
      __builtin_amdgcn_s_wait_tensorcnt(0);
    __syncthreads();
  }

  const float ax = __uint_as_float(scales[0]);
  const float aw = __uint_as_float(scales[1]);
  const float sc = (ax * aw) * (1.0f / 16129.0f);
  const int row0 = bm + wm * 64 + (lane >> 4) * 8;
  const int col0 = bn + wn * 32 + (lane & 15);
  #pragma unroll
  for (int ms = 0; ms < 4; ++ms)
    #pragma unroll
    for (int ns = 0; ns < 2; ++ns)
      #pragma unroll
      for (int r = 0; r < 8; ++r) {
        float vv = (float)acc[ms][ns][r] * sc;
        vv = fminf(fmaxf(vv, 0.0f), 6.0f);
        out[(size_t)(row0 + ms * 16 + r) * GN + col0 + ns * 16] = vv;
      }
}

extern "C" void kernel_launch(void* const* d_in, const int* in_sizes, int n_in,
                              void* d_out, int out_size, void* d_ws, size_t ws_size,
                              hipStream_t stream) {
  const float* x = (const float*)d_in[0];
  const float* W = (const float*)d_in[1];
  float* out = (float*)d_out;

  unsigned* scales = (unsigned*)d_ws;
  unsigned char* Qx  = (unsigned char*)d_ws + 1024;
  unsigned char* QwT = Qx + (size_t)GM * GK;

  init_scales_kernel<<<1, 32, 0, stream>>>(scales);

  absmax_kernel<<<2048, 256, 0, stream>>>(x, (size_t)GM * GK / 4, scales + 0);
  absmax_kernel<<<2048, 256, 0, stream>>>(W, (size_t)GN * GK / 4, scales + 1);

  const size_t n4x = (size_t)GM * GK / 4;
  quant_x_kernel<<<(unsigned)((n4x + 255) / 256), 256, 0, stream>>>(
      x, (unsigned*)Qx, scales + 0, n4x);

  quant_wT_kernel<<<dim3(GK / 64, GN / 64), 256, 0, stream>>>(W, QwT, scales + 1);

  gemm_i8_wmma_kernel<<<dim3(GM / 128, GN / 128), 256, SMEM_BYTES, stream>>>(
      Qx, QwT, out, scales);
}
